// ATDecoder_89472758710370
// MI455X (gfx1250) — matrix operand order repaired and checked by reading, not compiled, not run on hardware
//
#include <hip/hip_runtime.h>
#include <hip/hip_bf16.h>

// ---- types for WMMA ----
typedef __bf16  bf16_t;
typedef __attribute__((ext_vector_type(16))) __bf16 v16bf;
typedef __attribute__((ext_vector_type(8)))  float  v8f;

#define IN_CH 128
#define HID   64
#define NEG_SLOPE 0.01f

// =====================================================================
// Phase 1: per-node GEMM.
//   Atab[n][o] = sum_k z[n][k]*W1l[o][k] + b1l[o]
//   Btab[n][o] = sum_k z[n][k]*W1r[o][k] + b1r[o]
// One wave computes one 16-node tile across all 64 output channels for
// BOTH weight matrices: 2 weights * 4 n-tiles * 4 k-steps = 32 WMMAs.
// Weights are staged in LDS as bf16 once per block.
// =====================================================================
__global__ __launch_bounds__(256)
void atdec_node_gemm(const float* __restrict__ z,
                     const float* __restrict__ W1l,
                     const float* __restrict__ b1l,
                     const float* __restrict__ W1r,
                     const float* __restrict__ b1r,
                     float* __restrict__ Atab,
                     float* __restrict__ Btab,
                     int n_nodes, int n_tiles)
{
    __shared__ bf16_t wlds[2 * HID * IN_CH];   // 32 KB

    // ---- stage both weight matrices into LDS as bf16 ----
    for (int i = threadIdx.x; i < 2 * HID * IN_CH; i += blockDim.x) {
        float v = (i < HID * IN_CH) ? W1l[i] : W1r[i - HID * IN_CH];
        wlds[i] = (bf16_t)v;
    }
    __syncthreads();

    const int wave = threadIdx.x >> 5;
    const int lane = threadIdx.x & 31;
    const int tile = blockIdx.x * 8 + wave;
    if (tile >= n_tiles) return;

    const int row = lane & 15;          // M within tile (this lane's row)
    const int hi  = (lane >> 4) & 1;    // which K-half this lane holds
    const bool full_tile = (tile * 16 + 16) <= n_nodes;

    // ---- build 4 A fragments (K = 0..127 in steps of 32), bf16 ----
    // A packing (this code's contiguous k order, same as B): lanes<16 hold K = kk+[0..15],
    //                           lanes>=16 hold K = kk+[16..31].
    int m = tile * 16 + row;
    if (m >= n_nodes) m = n_nodes - 1;          // safe clamp (loads only)
    const float* zrow = z + (size_t)m * IN_CH;

    v16bf afrag[4];
    #pragma unroll
    for (int kk = 0; kk < 4; ++kk) {
        const int kbase = kk * 32 + hi * 16;
        const float4 f0 = *(const float4*)(zrow + kbase + 0);
        const float4 f1 = *(const float4*)(zrow + kbase + 4);
        const float4 f2 = *(const float4*)(zrow + kbase + 8);
        const float4 f3 = *(const float4*)(zrow + kbase + 12);
        afrag[kk][0]  = (bf16_t)f0.x; afrag[kk][1]  = (bf16_t)f0.y;
        afrag[kk][2]  = (bf16_t)f0.z; afrag[kk][3]  = (bf16_t)f0.w;
        afrag[kk][4]  = (bf16_t)f1.x; afrag[kk][5]  = (bf16_t)f1.y;
        afrag[kk][6]  = (bf16_t)f1.z; afrag[kk][7]  = (bf16_t)f1.w;
        afrag[kk][8]  = (bf16_t)f2.x; afrag[kk][9]  = (bf16_t)f2.y;
        afrag[kk][10] = (bf16_t)f2.z; afrag[kk][11] = (bf16_t)f2.w;
        afrag[kk][12] = (bf16_t)f3.x; afrag[kk][13] = (bf16_t)f3.y;
        afrag[kk][14] = (bf16_t)f3.z; afrag[kk][15] = (bf16_t)f3.w;
    }

    const int col = lane & 15;          // N within 16-wide tile

    #pragma unroll
    for (int w = 0; w < 2; ++w) {
        const float* bias  = (w == 0) ? b1l : b1r;
        float*       table = (w == 0) ? Atab : Btab;
        const bf16_t* wbase = wlds + w * HID * IN_CH;
        #pragma unroll
        for (int nt = 0; nt < 4; ++nt) {
            const int n0 = nt * 16;
            // B packing (same contiguous k order as A): lane holds column (lane&15),
            // lanes<16: K = kk+0..15, lanes>=16: K = kk+16..31, contiguous.
            const bf16_t* wp = wbase + (size_t)(n0 + col) * IN_CH + hi * 16;

            v8f acc = {};
            #pragma unroll
            for (int kk = 0; kk < 4; ++kk) {
                v16bf bfrag = *(const v16bf*)(wp + kk * 32);  // 2x ds_load_b128
                acc = __builtin_amdgcn_wmma_f32_16x16x32_bf16(
                        false, afrag[kk], false, bfrag,
                        (short)0, acc, false, false);
            }

            const float bv = bias[n0 + col];
            // D layout: VGPR r -> M = r + 8*hi, N = col
            float* tbase = table + (size_t)(tile * 16 + hi * 8) * HID + n0 + col;
            if (full_tile) {
                // branchless fast path: all 16 rows in range
                #pragma unroll
                for (int r = 0; r < 8; ++r)
                    tbase[(size_t)r * HID] = acc[r] + bv;
            } else {
                #pragma unroll
                for (int r = 0; r < 8; ++r) {
                    const int mm = tile * 16 + r + hi * 8;
                    if (mm < n_nodes)
                        tbase[(size_t)r * HID] = acc[r] + bv;
                }
            }
        }
    }
}

// =====================================================================
// Phase 2: per-edge decode. Tables are 51 MB -> L2-resident gathers.
//   out[e] = sigmoid( sum_o W2[o]*leaky(A[src][o]+B[dst][o]) + b2 )
// =====================================================================
__global__ __launch_bounds__(256)
void atdec_edge(const int* __restrict__ ei,
                const float* __restrict__ Atab,
                const float* __restrict__ Btab,
                const float* __restrict__ W2,
                const float* __restrict__ b2,
                float* __restrict__ out,
                int n_edges)
{
    const int e = blockIdx.x * blockDim.x + threadIdx.x;
    if (e >= n_edges) return;

    const int s = ei[e];
    const int d = ei[n_edges + e];

    const float4* a4 = (const float4*)(Atab + (size_t)s * HID);
    const float4* b4 = (const float4*)(Btab + (size_t)d * HID);
    const float4* w4 = (const float4*)W2;

    float acc = b2[0];
    #pragma unroll
    for (int j = 0; j < HID / 4; ++j) {
        const float4 a = a4[j];
        const float4 b = b4[j];
        const float4 w = w4[j];
        float hx = a.x + b.x; hx = (hx >= 0.0f) ? hx : NEG_SLOPE * hx;
        float hy = a.y + b.y; hy = (hy >= 0.0f) ? hy : NEG_SLOPE * hy;
        float hz = a.z + b.z; hz = (hz >= 0.0f) ? hz : NEG_SLOPE * hz;
        float hw = a.w + b.w; hw = (hw >= 0.0f) ? hw : NEG_SLOPE * hw;
        acc = fmaf(w.x, hx, acc);
        acc = fmaf(w.y, hy, acc);
        acc = fmaf(w.z, hz, acc);
        acc = fmaf(w.w, hw, acc);
    }
    out[e] = 1.0f / (1.0f + __expf(-acc));
}

// =====================================================================
// Launcher
// =====================================================================
extern "C" void kernel_launch(void* const* d_in, const int* in_sizes, int n_in,
                              void* d_out, int out_size, void* d_ws, size_t ws_size,
                              hipStream_t stream)
{
    const float* z   = (const float*)d_in[0];   // [N, 128]
    const int*   ei  = (const int*)  d_in[1];   // [2, E]
    const float* W1l = (const float*)d_in[2];   // [64, 128]
    const float* b1l = (const float*)d_in[3];   // [64]
    const float* W1r = (const float*)d_in[4];   // [64, 128]
    const float* b1r = (const float*)d_in[5];   // [64]
    const float* W2  = (const float*)d_in[6];   // [64]
    const float* b2  = (const float*)d_in[7];   // [1]
    float* out = (float*)d_out;

    const int n_nodes = in_sizes[0] / IN_CH;
    const int n_edges = in_sizes[1] / 2;

    float* Atab = (float*)d_ws;                       // [N, 64]
    float* Btab = Atab + (size_t)n_nodes * HID;       // [N, 64]

    const int n_tiles = (n_nodes + 15) / 16;
    const int gemm_blocks = (n_tiles + 7) / 8;        // 8 waves/block
    atdec_node_gemm<<<gemm_blocks, 256, 0, stream>>>(
        z, W1l, b1l, W1r, b1r, Atab, Btab, n_nodes, n_tiles);

    const int edge_blocks = (n_edges + 255) / 256;
    atdec_edge<<<edge_blocks, 256, 0, stream>>>(
        ei, Atab, Btab, W2, b2, out, n_edges);
}
